// DetailLayer_58488864637449
// MI455X (gfx1250) — compile-verified, matrix operand order checked by reading, not run on hardware
//
#include <hip/hip_runtime.h>
#include <hip/hip_bf16.h>

#define NPTS 800000
#define NVOX 150000
#define NG   1200
#define LPAD 160
#define DM   128
#define NH   4
#define HD   32
#define DFF  16
#define QKVD (3 * DM)   // 384

typedef __bf16 bf16_t;
typedef __attribute__((ext_vector_type(16))) __bf16 v16bf;
typedef __attribute__((ext_vector_type(8)))  float  v8f;
typedef __attribute__((ext_vector_type(4)))  unsigned int v4u;
typedef __attribute__((ext_vector_type(8)))  int v8i;
typedef __attribute__((ext_vector_type(4)))  int v4i;

// ---------------- WMMA helpers (layouts per CDNA5 ISA 7.12.2) ----------------

__device__ __forceinline__ int frag_k(int e, int half) {
    // 16-bit A-matrix 16x32: lanes 0-15 hold K=0..7 (v0..3), K=16..23 (v4..7);
    // lanes 16-31 hold K=8..15 and K=24..31.
    return e + half * 8 + ((e >= 8) ? 8 : 0);
}

// Load A (or B from a row-major [N,K] weight) fragment: row = lane&15, ld in elements.
__device__ __forceinline__ v16bf load_frag_row(const bf16_t* base, int ld) {
    int lane = threadIdx.x & 31;
    int half = lane >> 4, r = lane & 15;
    const bf16_t* p = base + (size_t)r * ld;
    v16bf a;
#pragma unroll
    for (int e = 0; e < 16; ++e) a[e] = p[frag_k(e, half)];
    return a;
}

// Load B fragment where B[k,n] comes from a row-major [K, ld] source (column gather).
__device__ __forceinline__ v16bf load_frag_col(const bf16_t* base, int ld) {
    int lane = threadIdx.x & 31;
    int half = lane >> 4, c = lane & 15;
    v16bf b;
#pragma unroll
    for (int e = 0; e < 16; ++e) b[e] = base[(size_t)frag_k(e, half) * ld + c];
    return b;
}

__device__ __forceinline__ v8f wmma_bf16(v16bf a, v16bf b, v8f c) {
    // (neg_a, A, neg_b, B, c_mod, C, reuse_a, reuse_b)
    return __builtin_amdgcn_wmma_f32_16x16x32_bf16(false, a, false, b, (short)0, c,
                                                   false, false);
}

// ---------------- TDM: 1-D tile DMA global -> LDS (CDNA5 ISA ch.8 D#) -------
// nelem bf16 elements, contiguous. Issued by the calling wave (EXEC ignored).

__device__ __forceinline__ void tdm_load_1d(unsigned lds_off, const void* gaddr,
                                            unsigned nelem) {
    unsigned long long ga = (unsigned long long)gaddr;
    v4u g0;
    g0[0] = 1u;                                        // count=1, user descriptor
    g0[1] = lds_off;                                   // lds_addr (bytes)
    g0[2] = (unsigned)(ga & 0xffffffffu);              // global_addr[31:0]
    g0[3] = (unsigned)((ga >> 32) & 0x01ffffffu)       // global_addr[56:32]
          | (2u << 30);                                // type = 2 ("image")
    v8i g1;
    g1[0] = (int)(1u << 16);                           // data_size = 1 -> 2 bytes
    g1[1] = (int)((nelem & 0xffffu) << 16);            // tensor_dim0[15:0]
    g1[2] = (int)((nelem >> 16) & 0xffffu)             // tensor_dim0[31:16]
          | (int)(1u << 16);                           // tensor_dim1 = 1
    g1[3] = (int)((nelem & 0xffffu) << 16);            // tile_dim0 (16-bit)
    g1[4] = 0;                                         // tile_dim1/2 unused
    g1[5] = (int)nelem;                                // tensor_dim0_stride[31:0]
    g1[6] = 0;
    g1[7] = 0;
    v4i z4 = {0, 0, 0, 0};
#if __clang_major__ >= 23
    v8i z8 = {0, 0, 0, 0, 0, 0, 0, 0};
    __builtin_amdgcn_tensor_load_to_lds(g0, g1, z4, z4, z8, 0);
#else
    __builtin_amdgcn_tensor_load_to_lds(g0, g1, z4, z4, 0);
#endif
}

// ---------------- Stage kernels ----------------

__global__ void prep_weights_kernel(const float* wqkv, const float* wout,
                                    const float* w1, const float* w2,
                                    bf16_t* wqkvb, bf16_t* woutb,
                                    bf16_t* w1b, bf16_t* w2pb) {
    int i = blockIdx.x * blockDim.x + threadIdx.x;
    if (i < 3 * DM * DM) wqkvb[i] = (bf16_t)wqkv[i];
    if (i < DM * DM)     woutb[i] = (bf16_t)wout[i];
    if (i < DFF * DM)    w1b[i]   = (bf16_t)w1[i];
    if (i < DM * 32) {   // w2 is [DM, DFF]; pad K 16 -> 32 with zeros
        int r = i / 32, c = i % 32;
        w2pb[i] = (bf16_t)((c < DFF) ? w2[r * DFF + c] : 0.0f);
    }
}

__global__ __launch_bounds__(128) void scatter_kernel(const float* pts, const int* unq,
                                                      const int* big, float* sums,
                                                      float* cnt, int* vg) {
    int i = blockIdx.x;          // point id
    int d = threadIdx.x;         // channel
    int v = unq[i];
    atomicAdd(&sums[(size_t)v * DM + d], pts[(size_t)i * DM + d]);
    if (d == 0) {
        atomicAdd(&cnt[v], 1.0f);
        vg[v] = big[i];          // same value for all points of a voxel
    }
}

__global__ void means_kernel(float* sums, const float* cnt) {
    size_t i = (size_t)blockIdx.x * blockDim.x + threadIdx.x;
    if (i < (size_t)NVOX * DM) {
        float c = cnt[i / DM];
        sums[i] = sums[i] / fmaxf(c, 1.0f);
    }
}

// Deterministic ragged->dense: per group, scan voxels in ascending id, rank = slot.
__global__ __launch_bounds__(256) void pack_kernel(const int* voxel_group, int* glist,
                                                   int* n_valid) {
    int g = blockIdx.x;
    __shared__ int warpCnt[8];
    __shared__ int runningS;
    if (threadIdx.x == 0) runningS = 0;
    __syncthreads();
    int lane = threadIdx.x & 31, w = threadIdx.x >> 5;
    for (int base = 0; base < NVOX; base += 256) {
        int i = base + (int)threadIdx.x;
        int flag = (i < NVOX) && (voxel_group[i] == g);
        unsigned int mm = (unsigned int)__ballot(flag);
        int mypre = __popc(mm & ((1u << lane) - 1u));
        if (lane == 0) warpCnt[w] = __popc(mm);
        __syncthreads();
        int woff = 0;
#pragma unroll
        for (int j = 0; j < 8; ++j) woff += (j < w) ? warpCnt[j] : 0;
        int chunkTot = 0;
#pragma unroll
        for (int j = 0; j < 8; ++j) chunkTot += warpCnt[j];
        int slot = runningS + woff + mypre;
        if (flag && slot < LPAD) glist[g * LPAD + slot] = i;
        __syncthreads();
        if (threadIdx.x == 0) runningS += chunkTot;
        __syncthreads();
    }
    if (threadIdx.x == 0) n_valid[g] = (runningS < LPAD) ? runningS : LPAD;
}

__global__ __launch_bounds__(128) void gather_kernel(const float* means, const int* glist,
                                                     const int* n_valid, float* feats,
                                                     bf16_t* featsb) {
    int row = blockIdx.x;              // 0 .. G*L-1
    int g = row / LPAD, l = row % LPAD;
    int d = threadIdx.x;
    float val = 0.0f;
    if (l < n_valid[g]) {
        int v = glist[row];
        val = means[(size_t)v * DM + d];
    }
    feats[(size_t)row * DM + d]  = val;
    featsb[(size_t)row * DM + d] = (bf16_t)val;
}

// QKV projection: per group [160x128] x [128x384]^T, bf16 WMMA, fp32 accum.
// Output fused [G, L, 384] (cols 0..127=Q, 128..255=K, 256..383=V).
__global__ __launch_bounds__(256) void qkv_kernel(const bf16_t* featsb, const bf16_t* wqkv,
                                                  const float* bias, bf16_t* qkvb) {
    int g = blockIdx.x;
    __shared__ bf16_t s[LPAD * DM];    // 40 KB, filled by TDM
    const bf16_t* src = featsb + (size_t)g * LPAD * DM;
    if (threadIdx.x < 32) {            // wave 0 issues the DMA
        tdm_load_1d((unsigned)(unsigned long long)&s[0], src, LPAD * DM);
        __builtin_amdgcn_s_wait_tensorcnt(0);
    }
    __syncthreads();
    int wave = threadIdx.x >> 5;
    int lane = threadIdx.x & 31, half = lane >> 4, col = lane & 15;
    for (int t = wave; t < 240; t += 8) {      // 10 M-tiles x 24 N-tiles
        int mt = t / 24, nt = t % 24;
        v8f c = {};
#pragma unroll
        for (int kk = 0; kk < 4; ++kk) {
            v16bf a = load_frag_row(s + mt * 16 * DM + kk * 32, DM);
            v16bf b = load_frag_row(wqkv + nt * 16 * DM + kk * 32, DM);
            c = wmma_bf16(a, b, c);
        }
        int gcol = nt * 16 + col;
        float bi = bias[gcol];
        bf16_t* dst = qkvb + ((size_t)g * LPAD + mt * 16) * QKVD + gcol;
#pragma unroll
        for (int j = 0; j < 8; ++j) dst[(j + 8 * half) * QKVD] = (bf16_t)(c[j] + bi);
    }
}

// Attention: one wave per head; streaming over 10 M-tiles; masked softmax in LDS.
__global__ __launch_bounds__(128) void attn_kernel(const bf16_t* qkvb, const int* n_valid,
                                                   bf16_t* ctx) {
    int g = blockIdx.x;
    int wave = threadIdx.x >> 5, lane = threadIdx.x & 31;
    int h = wave;
    __shared__ float  sc[NH][16 * LPAD];   // 40 KB
    __shared__ bf16_t pb[NH][16 * LPAD];   // 20 KB
    int nv = n_valid[g];
    const bf16_t* base = qkvb + (size_t)g * LPAD * QKVD;
    const bf16_t* Q  = base + h * HD;              // cols   0..127
    const bf16_t* K  = base + DM + h * HD;         // cols 128..255
    const bf16_t* Vv = base + 2 * DM + h * HD;     // cols 256..383
    const float scale = 0.17677669529663687f;      // 1/sqrt(32)
    int half = lane >> 4, col = lane & 15;
    for (int mt = 0; mt < 10; ++mt) {
        v16bf aq = load_frag_row(Q + (size_t)(mt * 16) * QKVD, QKVD);
        for (int nt = 0; nt < 10; ++nt) {
            v16bf bk = load_frag_row(K + (size_t)(nt * 16) * QKVD, QKVD);
            v8f c = {};
            c = wmma_bf16(aq, bk, c);
#pragma unroll
            for (int j = 0; j < 8; ++j)
                sc[h][(j + 8 * half) * LPAD + nt * 16 + col] = c[j] * scale;
        }
        __syncthreads();
        if (lane < 16) {
            float* row = &sc[h][lane * LPAD];
            float mx = -3.4e38f;
            for (int n = 0; n < nv; ++n) mx = fmaxf(mx, row[n]);
            float sum = 0.0f;
            for (int n = 0; n < nv; ++n) { float e = __expf(row[n] - mx); row[n] = e; sum += e; }
            float inv = 1.0f / sum;
            bf16_t* prow = &pb[h][lane * LPAD];
            for (int n = 0; n < LPAD; ++n)
                prow[n] = (bf16_t)((n < nv) ? row[n] * inv : 0.0f);
        }
        __syncthreads();
#pragma unroll
        for (int nd = 0; nd < 2; ++nd) {
            v8f c = {};
#pragma unroll
            for (int kk = 0; kk < 5; ++kk) {       // K = 160 = 5 x 32
                v16bf a = load_frag_row(&pb[h][kk * 32], LPAD);
                v16bf b = load_frag_col(Vv + (size_t)(kk * 32) * QKVD + nd * 16, QKVD);
                c = wmma_bf16(a, b, c);
            }
            bf16_t* dst = ctx + (size_t)g * LPAD * DM + (size_t)(mt * 16) * DM
                        + h * HD + nd * 16 + col;
#pragma unroll
            for (int j = 0; j < 8; ++j) dst[(j + 8 * half) * DM] = (bf16_t)c[j];
        }
        __syncthreads();
    }
}

// Out-proj + bias + residual (writes pre-LN x into feats, in place per element).
__global__ __launch_bounds__(256) void outproj_kernel(const bf16_t* ctx, const bf16_t* wout,
                                                      const float* bias, float* xf) {
    int g = blockIdx.x;
    __shared__ bf16_t s[LPAD * DM];    // 40 KB, filled by TDM
    const bf16_t* src = ctx + (size_t)g * LPAD * DM;
    if (threadIdx.x < 32) {
        tdm_load_1d((unsigned)(unsigned long long)&s[0], src, LPAD * DM);
        __builtin_amdgcn_s_wait_tensorcnt(0);
    }
    __syncthreads();
    int wave = threadIdx.x >> 5;
    int lane = threadIdx.x & 31, half = lane >> 4, col = lane & 15;
    for (int t = wave; t < 80; t += 8) {       // 10 x 8 tiles
        int mt = t / 8, nt = t % 8;
        v8f c = {};
#pragma unroll
        for (int kk = 0; kk < 4; ++kk) {
            v16bf a = load_frag_row(s + mt * 16 * DM + kk * 32, DM);
            v16bf b = load_frag_row(wout + nt * 16 * DM + kk * 32, DM);
            c = wmma_bf16(a, b, c);
        }
        int gcol = nt * 16 + col;
        float bi = bias[gcol];
        float* dst = xf + (size_t)g * LPAD * DM + (size_t)(mt * 16) * DM + gcol;
#pragma unroll
        for (int j = 0; j < 8; ++j) {
            float r = dst[(j + 8 * half) * DM];
            dst[(j + 8 * half) * DM] = r + c[j] + bi;
        }
    }
}

__global__ __launch_bounds__(128) void ln_kernel(const float* in, const float* gg,
                                                 const float* bb, float* outf,
                                                 bf16_t* outb) {
    int row = blockIdx.x, d = threadIdx.x;
    __shared__ float red[DM];
    float v = in[(size_t)row * DM + d];
    red[d] = v;
    __syncthreads();
    for (int s = 64; s > 0; s >>= 1) { if (d < s) red[d] += red[d + s]; __syncthreads(); }
    float mu = red[0] * (1.0f / DM);
    __syncthreads();
    float diff = v - mu;
    red[d] = diff * diff;
    __syncthreads();
    for (int s = 64; s > 0; s >>= 1) { if (d < s) red[d] += red[d + s]; __syncthreads(); }
    float var = red[0] * (1.0f / DM);
    float r = diff * rsqrtf(var + 1e-5f) * gg[d] + bb[d];
    if (outf) outf[(size_t)row * DM + d] = r;
    if (outb) outb[(size_t)row * DM + d] = (bf16_t)r;
}

// FFN1: hidden = relu(x @ W1^T + b1), stored bf16 [G, L, 32] (cols 16..31 zero).
__global__ __launch_bounds__(64) void ffn1_kernel(const bf16_t* xb, const bf16_t* w1b,
                                                  const float* b1, bf16_t* hid) {
    int g = blockIdx.x;
    int wave = threadIdx.x >> 5, lane = threadIdx.x & 31;
    int half = lane >> 4, col = lane & 15;
    const bf16_t* src = xb + (size_t)g * LPAD * DM;
    for (int mt = wave; mt < 10; mt += 2) {
        v8f c = {};
#pragma unroll
        for (int kk = 0; kk < 4; ++kk) {
            v16bf a = load_frag_row(src + (size_t)(mt * 16) * DM + kk * 32, DM);
            v16bf b = load_frag_row(w1b + kk * 32, DM);   // 16 rows x 128
            c = wmma_bf16(a, b, c);
        }
        float bi = b1[col];
        bf16_t* dst = hid + (size_t)g * LPAD * 32 + (size_t)(mt * 16) * 32;
#pragma unroll
        for (int j = 0; j < 8; ++j) {
            int r = j + 8 * half;
            dst[r * 32 + col]      = (bf16_t)fmaxf(c[j] + bi, 0.0f);
            dst[r * 32 + 16 + col] = (bf16_t)0.0f;
        }
    }
}

// FFN2: y = hidden @ W2^T + b2 + x, written into feats (pre-LN2).
__global__ __launch_bounds__(256) void ffn2_kernel(const bf16_t* hid, const bf16_t* w2p,
                                                   const float* b2, float* xf) {
    int g = blockIdx.x;
    int wave = threadIdx.x >> 5;
    int lane = threadIdx.x & 31, half = lane >> 4, col = lane & 15;
    for (int t = wave; t < 80; t += 8) {
        int mt = t / 8, nt = t % 8;
        v16bf a = load_frag_row(hid + (size_t)g * LPAD * 32 + (size_t)(mt * 16) * 32, 32);
        v16bf b = load_frag_row(w2p + nt * 16 * 32, 32);
        v8f c = {};
        c = wmma_bf16(a, b, c);
        int gcol = nt * 16 + col;
        float bi = b2[gcol];
        float* dst = xf + (size_t)g * LPAD * DM + (size_t)(mt * 16) * DM + gcol;
#pragma unroll
        for (int j = 0; j < 8; ++j) {
            float r = dst[(j + 8 * half) * DM];
            dst[(j + 8 * half) * DM] = r + c[j] + bi;
        }
    }
}

// ---------------- Launch ----------------

extern "C" void kernel_launch(void* const* d_in, const int* in_sizes, int n_in,
                              void* d_out, int out_size, void* d_ws, size_t ws_size,
                              hipStream_t stream) {
    const float* points   = (const float*)d_in[0];
    const int*   unq_inv  = (const int*)d_in[1];
    const int*   big_idx  = (const int*)d_in[2];
    const float* in_w     = (const float*)d_in[3];
    const float* in_b     = (const float*)d_in[4];
    const float* out_w    = (const float*)d_in[5];
    const float* out_b    = (const float*)d_in[6];
    const float* w1       = (const float*)d_in[7];
    const float* b1       = (const float*)d_in[8];
    const float* w2       = (const float*)d_in[9];
    const float* b2       = (const float*)d_in[10];
    const float* ln1_g    = (const float*)d_in[11];
    const float* ln1_b    = (const float*)d_in[12];
    const float* ln2_g    = (const float*)d_in[13];
    const float* ln2_b    = (const float*)d_in[14];
    float* out = (float*)d_out;

    char* ws = (char*)d_ws;
    size_t off = 0;
    auto alloc = [&](size_t bytes) -> size_t {
        size_t o = off;
        off = (off + bytes + 255) & ~(size_t)255;
        return o;
    };
    const size_t GLD = (size_t)NG * LPAD * DM;
    size_t o_sums  = alloc((size_t)NVOX * DM * 4);
    size_t o_cnt   = alloc((size_t)NVOX * 4);
    size_t o_vg    = alloc((size_t)NVOX * 4);
    size_t o_glist = alloc((size_t)NG * LPAD * 4);
    size_t o_nval  = alloc((size_t)NG * 4);
    size_t o_feats = alloc(GLD * 4);
    size_t o_fb    = alloc(GLD * 2);
    size_t o_qkv   = alloc(GLD * 3 * 2);
    size_t o_ctx   = alloc(GLD * 2);
    size_t o_hid   = alloc((size_t)NG * LPAD * 32 * 2);
    size_t o_wqkv  = alloc((size_t)3 * DM * DM * 2);
    size_t o_wout  = alloc((size_t)DM * DM * 2);
    size_t o_w1    = alloc((size_t)DFF * DM * 2);
    size_t o_w2p   = alloc((size_t)DM * 32 * 2);

    float*  sums   = (float*)(ws + o_sums);
    float*  cnt    = (float*)(ws + o_cnt);
    int*    vg     = (int*)(ws + o_vg);
    int*    glist  = (int*)(ws + o_glist);
    int*    nval   = (int*)(ws + o_nval);
    float*  feats  = (float*)(ws + o_feats);
    bf16_t* fb     = (bf16_t*)(ws + o_fb);
    bf16_t* qkvb   = (bf16_t*)(ws + o_qkv);
    bf16_t* ctx    = (bf16_t*)(ws + o_ctx);
    bf16_t* hid    = (bf16_t*)(ws + o_hid);
    bf16_t* wqkvb  = (bf16_t*)(ws + o_wqkv);
    bf16_t* woutb  = (bf16_t*)(ws + o_wout);
    bf16_t* w1b    = (bf16_t*)(ws + o_w1);
    bf16_t* w2pb   = (bf16_t*)(ws + o_w2p);

    // Zero accumulators; set voxel_group to -1 (empty voxels excluded from packing).
    hipMemsetAsync(sums, 0, (size_t)NVOX * DM * 4 + (size_t)NVOX * 4, stream);
    hipMemsetAsync(vg, 0xFF, (size_t)NVOX * 4, stream);

    prep_weights_kernel<<<(3 * DM * DM + 255) / 256, 256, 0, stream>>>(
        in_w, out_w, w1, w2, wqkvb, woutb, w1b, w2pb);

    scatter_kernel<<<NPTS, 128, 0, stream>>>(points, unq_inv, big_idx, sums, cnt, vg);
    means_kernel<<<((size_t)NVOX * DM + 255) / 256, 256, 0, stream>>>(sums, cnt);
    pack_kernel<<<NG, 256, 0, stream>>>(vg, glist, nval);
    gather_kernel<<<NG * LPAD, 128, 0, stream>>>(sums, glist, nval, feats, fb);

    qkv_kernel<<<NG, 256, 0, stream>>>(fb, wqkvb, in_b, qkvb);
    attn_kernel<<<NG, 128, 0, stream>>>(qkvb, nval, ctx);
    outproj_kernel<<<NG, 256, 0, stream>>>(ctx, woutb, out_b, feats);
    ln_kernel<<<NG * LPAD, 128, 0, stream>>>(feats, ln1_g, ln1_b, feats, fb);

    ffn1_kernel<<<NG, 64, 0, stream>>>(fb, w1b, b1, hid);
    ffn2_kernel<<<NG, 256, 0, stream>>>(hid, w2pb, b2, feats);
    ln_kernel<<<NG * LPAD, 128, 0, stream>>>(feats, ln2_g, ln2_b, out, (bf16_t*)nullptr);

    (void)in_sizes; (void)n_in; (void)out_size; (void)ws_size;
}
